// TransformerDecoder_69209103007791
// MI455X (gfx1250) — matrix operand order repaired and checked by reading, not compiled, not run on hardware
//
#include <hip/hip_runtime.h>
#include <hip/hip_bf16.h>

// ---------------------------------------------------------------------------
// Types for CDNA5 WMMA bf16 path
// ---------------------------------------------------------------------------
typedef __attribute__((ext_vector_type(16))) __bf16    v16bf;
typedef __attribute__((ext_vector_type(8)))  float     v8f;
typedef __attribute__((ext_vector_type(8)))  unsigned  v8u;

__device__ __forceinline__ unsigned short f2bf(float f) {
  union { float f; unsigned u; } x; x.f = f;
  unsigned u = x.u;
  u += 0x7fffu + ((u >> 16) & 1u);   // round-to-nearest-even to bf16
  return (unsigned short)(u >> 16);
}
__device__ __forceinline__ unsigned pk2bf(float lo, float hi) {
  return (unsigned)f2bf(lo) | ((unsigned)f2bf(hi) << 16);
}

// A-fragment (16x32 bf16): lane = M row; element i -> k = i + 16*half (contiguous)
__device__ __forceinline__ v16bf afrag_from_lds(const float* __restrict__ row, int half) {
  const float* lo = row + half * 16;
  const float* hi = row + half * 16 + 8;
  v8u p;
  p[0] = pk2bf(lo[0], lo[1]); p[1] = pk2bf(lo[2], lo[3]);
  p[2] = pk2bf(lo[4], lo[5]); p[3] = pk2bf(lo[6], lo[7]);
  p[4] = pk2bf(hi[0], hi[1]); p[5] = pk2bf(hi[2], hi[3]);
  p[6] = pk2bf(hi[4], hi[5]); p[7] = pk2bf(hi[6], hi[7]);
  return __builtin_bit_cast(v16bf, p);
}
// B-fragment (32x16 bf16): lane = N col; element i -> k = i + 16*half (contiguous)
__device__ __forceinline__ v16bf bfrag_from_lds(const float* __restrict__ row, int half) {
  const float* s = row + half * 16;
  v8u p;
  #pragma unroll
  for (int j = 0; j < 8; j++) p[j] = pk2bf(s[2 * j], s[2 * j + 1]);
  return __builtin_bit_cast(v16bf, p);
}

// ---------------------------------------------------------------------------
// Async tile fill: 128 rows x 32 fp32 (16 KB) via GLOBAL_LOAD_ASYNC_TO_LDS_B128
// (ASYNCcnt-tracked, no VGPR staging). 256 threads x 4 chunks of 16B.
// ---------------------------------------------------------------------------
__device__ __forceinline__ void async_fill_tile(const float* __restrict__ g, int ldk,
                                                unsigned ldsbase, int tid) {
  #pragma unroll
  for (int j = 0; j < 4; j++) {
    const int ci = j * 256 + tid;         // 0..1023 16B chunks
    const int r  = ci >> 3;               // row 0..127
    const int c4 = (ci & 7) << 2;         // float col 0,4,..,28
    const float* gp = g + (size_t)r * ldk + c4;
    const unsigned lp = ldsbase + (unsigned)((r * 32 + c4) * 4);
    asm volatile("global_load_async_to_lds_b128 %0, %1, off"
                 :: "v"(lp), "v"(gp) : "memory");
  }
}
__device__ __forceinline__ void wait_async0() {
  asm volatile("s_wait_asynccnt 0x0" ::: "memory");
}

// ---------------------------------------------------------------------------
// GEMM: C[M,N] = A[M,K] @ W[N,K]^T (+ bias[N]) (+ReLU)
// Block = 256 threads (8 waves). Tile 128x128, BK = 32, double-buffered LDS
// filled by async-to-LDS; wave = 32x64 patch -> 8 v_wmma_f32_16x16x32_bf16
// per K-step. M,N multiples of 128; K multiple of 32.
// ---------------------------------------------------------------------------
#define TM 128
#define TN 128
#define TK 32

__global__ __launch_bounds__(256) void k_gemm(
    const float* __restrict__ A, const float* __restrict__ W,
    const float* __restrict__ bias, float* __restrict__ C,
    int M, int N, int K, int relu)
{
  __shared__ float sA[2][TM * TK];   // fp32 tiles (async copied raw)
  __shared__ float sB[2][TN * TK];

  const int tid  = threadIdx.x;
  const int wave = tid >> 5;
  const int lane = tid & 31;
  const int half = lane >> 4;
  const int l16  = lane & 15;

  const int m0 = blockIdx.y * TM;
  const int n0 = blockIdx.x * TN;
  const int wr = (wave >> 1) * 32;   // wave row offset (4 waves x 32 rows)
  const int wc = (wave & 1) * 64;    // wave col offset (2 waves x 64 cols)

  const unsigned baseA[2] = { (unsigned)(uintptr_t)&sA[0][0], (unsigned)(uintptr_t)&sA[1][0] };
  const unsigned baseB[2] = { (unsigned)(uintptr_t)&sB[0][0], (unsigned)(uintptr_t)&sB[1][0] };
  const float* gA = A + (size_t)m0 * K;
  const float* gW = W + (size_t)n0 * K;

  v8f acc[2][4] = {};

  const int steps = K >> 5;
  // prologue: fill buffer 0 with K-step 0
  async_fill_tile(gA, K, baseA[0], tid);
  async_fill_tile(gW, K, baseB[0], tid);

  for (int kt = 0; kt < steps; kt++) {
    const int cur = kt & 1;
    wait_async0();        // current tile landed in LDS (next not yet issued)
    __syncthreads();

    // overlap: issue next tile's async fill before the matrix work
    if (kt + 1 < steps) {
      const int k1 = (kt + 1) << 5;
      async_fill_tile(gA + k1, K, baseA[cur ^ 1], tid);
      async_fill_tile(gW + k1, K, baseB[cur ^ 1], tid);
    }

    // fragments (fp32 LDS -> bf16 packs) + 8 WMMA
    const float* tA = &sA[cur][0];
    const float* tB = &sB[cur][0];
    v16bf af[2], bf[4];
    #pragma unroll
    for (int i = 0; i < 2; i++)
      af[i] = afrag_from_lds(tA + (size_t)(wr + i * 16 + l16) * TK, half);
    #pragma unroll
    for (int j = 0; j < 4; j++)
      bf[j] = bfrag_from_lds(tB + (size_t)(wc + j * 16 + l16) * TK, half);

    #pragma unroll
    for (int i = 0; i < 2; i++)
      #pragma unroll
      for (int j = 0; j < 4; j++)
        acc[i][j] = __builtin_amdgcn_wmma_f32_16x16x32_bf16(
            false, af[i], false, bf[j], (short)0, acc[i][j], false, false);

    __syncthreads();      // all waves done reading cur before it is refilled
  }

  // store: D layout -> row = base + r + 8*half, col = base + l16
  #pragma unroll
  for (int i = 0; i < 2; i++) {
    const int rowbase = m0 + wr + i * 16 + half * 8;
    #pragma unroll
    for (int j = 0; j < 4; j++) {
      const int col = n0 + wc + j * 16 + l16;
      const float bv = bias ? bias[col] : 0.0f;
      #pragma unroll
      for (int r = 0; r < 8; r++) {
        float v = acc[i][j][r] + bv;
        if (relu) v = fmaxf(v, 0.0f);
        C[(size_t)(rowbase + r) * N + col] = v;
      }
    }
  }
}

// ---------------------------------------------------------------------------
// Attention: one block (256 thr) per (b, h, q) row. dk = 64, H = 8, D = 512.
// mask: key masked if kseq[b*Lk+j]==0 (PAD) or (causal && j>q).
// ---------------------------------------------------------------------------
__global__ __launch_bounds__(256) void k_attn(
    const float* __restrict__ Q, const float* __restrict__ Kb,
    const float* __restrict__ Vb, float* __restrict__ O,
    const int* __restrict__ kseq, int Lq, int Lk, int causal)
{
  __shared__ float p[512];
  __shared__ float red[256];
  __shared__ float qv[64];

  const int t = threadIdx.x;
  const int qidx = blockIdx.x;
  const int q = qidx % Lq;
  const int h = (qidx / Lq) % 8;
  const int b = qidx / (Lq * 8);

  const float* qrow = Q + ((size_t)(b * Lq + q)) * 512 + h * 64;
  if (t < 64) qv[t] = qrow[t];
  __syncthreads();

  float lmax = -1e30f;
  for (int j = t; j < Lk; j += 256) {
    const float* krow = Kb + ((size_t)(b * Lk + j)) * 512 + h * 64;
    float s = 0.0f;
    #pragma unroll 16
    for (int d = 0; d < 64; d++) s += qv[d] * krow[d];
    s *= 0.125f;   // 1/sqrt(64)
    const bool masked = (kseq[b * Lk + j] == 0) || (causal && (j > q));
    if (masked) s = -1e30f;
    p[j] = s;
    lmax = fmaxf(lmax, s);
  }
  red[t] = lmax; __syncthreads();
  for (int s = 128; s > 0; s >>= 1) {
    if (t < s) red[t] = fmaxf(red[t], red[t + s]);
    __syncthreads();
  }
  const float mx = red[0]; __syncthreads();

  float lsum = 0.0f;
  for (int j = t; j < Lk; j += 256) {
    const float e = __expf(p[j] - mx);
    p[j] = e; lsum += e;
  }
  red[t] = lsum; __syncthreads();
  for (int s = 128; s > 0; s >>= 1) {
    if (t < s) red[t] += red[t + s];
    __syncthreads();
  }
  const float inv = 1.0f / red[0]; __syncthreads();

  const int d = t & 63, part = t >> 6;
  float acc = 0.0f;
  for (int j = part; j < Lk; j += 4)
    acc += p[j] * Vb[((size_t)(b * Lk + j)) * 512 + h * 64 + d];
  red[t] = acc; __syncthreads();
  if (t < 64) {
    const float o = (red[t] + red[t + 64] + red[t + 128] + red[t + 192]) * inv;
    O[((size_t)(b * Lq + q)) * 512 + h * 64 + d] = o;
  }
}

// ---------------------------------------------------------------------------
// Fused residual + LayerNorm + padding-row zero. D = 512, block = 256.
// ---------------------------------------------------------------------------
__global__ __launch_bounds__(256) void k_res_ln(
    const float* __restrict__ X, const float* __restrict__ R,
    const float* __restrict__ gamma, const float* __restrict__ beta,
    const int* __restrict__ seq, float* __restrict__ Out)
{
  __shared__ float red[256];
  const int row = blockIdx.x;
  const int t = threadIdx.x;
  const size_t base = (size_t)row * 512;

  const float v0 = X[base + t]       + R[base + t];
  const float v1 = X[base + t + 256] + R[base + t + 256];

  red[t] = v0 + v1; __syncthreads();
  for (int s = 128; s > 0; s >>= 1) { if (t < s) red[t] += red[t + s]; __syncthreads(); }
  const float mean = red[0] * (1.0f / 512.0f); __syncthreads();

  const float d0 = v0 - mean, d1 = v1 - mean;
  red[t] = d0 * d0 + d1 * d1; __syncthreads();
  for (int s = 128; s > 0; s >>= 1) { if (t < s) red[t] += red[t + s]; __syncthreads(); }
  const float invs = rsqrtf(red[0] * (1.0f / 512.0f) + 1e-5f);

  const float m = (seq && seq[row] == 0) ? 0.0f : 1.0f;
  Out[base + t]       = (gamma[t]       * d0 * invs + beta[t])       * m;
  Out[base + t + 256] = (gamma[t + 256] * d1 * invs + beta[t + 256]) * m;
}

// ---------------------------------------------------------------------------
// Embedding: emb[tok] + sinusoidal PE (+ act_vecs @ act_w.T for decoder)
// ---------------------------------------------------------------------------
__global__ __launch_bounds__(256) void k_embed(
    const int* __restrict__ seq, const float* __restrict__ emb,
    const float* __restrict__ av, const float* __restrict__ aw,
    float* __restrict__ out, int L)
{
  const int row = blockIdx.x;
  const int pos = row % L;
  const int b = row / L;
  const int tok = seq[row];
  const float c = -logf(10000.0f) / 512.0f;
  for (int d = threadIdx.x; d < 512; d += 256) {
    const int i2 = (d >> 1) * 2;
    const float ang = (float)pos * __expf(c * (float)i2);
    const float pe = (d & 1) ? __cosf(ang) : __sinf(ang);
    float a = 0.0f;
    if (av) {
      const float* avr = av + (size_t)b * 128;
      const float* awr = aw + (size_t)d * 128;
      #pragma unroll 16
      for (int j = 0; j < 128; j++) a += avr[j] * awr[j];
    }
    out[(size_t)row * 512 + d] = emb[(size_t)tok * 512 + d] + pe + a;
  }
}

// ---------------------------------------------------------------------------
// Host orchestration
// ---------------------------------------------------------------------------
extern "C" void kernel_launch(void* const* d_in, const int* in_sizes, int n_in,
                              void* d_out, int out_size, void* d_ws, size_t ws_size,
                              hipStream_t stream)
{
  const int*   tgt_seq    = (const int*)  d_in[0];
  const int*   src_seq    = (const int*)  d_in[1];
  const float* act_vecs   = (const float*)d_in[2];
  const float* emb        = (const float*)d_in[3];
  const float* act_w      = (const float*)d_in[4];
  const float* enc_qkv_w  = (const float*)d_in[5];
  const float* enc_qkv_b  = (const float*)d_in[6];
  const float* enc_fc_w   = (const float*)d_in[7];
  const float* enc_fc_b   = (const float*)d_in[8];
  const float* enc_ffn_w1 = (const float*)d_in[9];
  const float* enc_ffn_b1 = (const float*)d_in[10];
  const float* enc_ffn_w2 = (const float*)d_in[11];
  const float* enc_ffn_b2 = (const float*)d_in[12];
  const float* enc_ln     = (const float*)d_in[13];
  const float* dec_qkv_w  = (const float*)d_in[14];
  const float* dec_qkv_b  = (const float*)d_in[15];
  const float* dec_fc_w   = (const float*)d_in[16];
  const float* dec_fc_b   = (const float*)d_in[17];
  const float* dec_ffn_w1 = (const float*)d_in[18];
  const float* dec_ffn_b1 = (const float*)d_in[19];
  const float* dec_ffn_w2 = (const float*)d_in[20];
  const float* dec_ffn_b2 = (const float*)d_in[21];
  const float* dec_ln     = (const float*)d_in[22];
  const float* proj_w     = (const float*)d_in[23];
  float* out = (float*)d_out;
  (void)in_sizes; (void)n_in; (void)out_size; (void)ws_size;

  constexpr int Dm = 512, Hh = 8, Dh = 2048, Ls = 512, Lt = 128, Bb = 8, NL = 6, Vv = 32000;
  const int Menc = Bb * Ls;   // 4096
  const int Mdec = Bb * Lt;   // 1024

  // workspace carve (~82 MB)
  float* ws = (float*)d_ws;
  size_t off = 0;
  auto carve = [&](size_t n) { float* p = ws + off; off += n; return p; };
  float* enc_x = carve((size_t)Menc * Dm);
  float* bq    = carve((size_t)Menc * Dm);
  float* bk    = carve((size_t)Menc * Dm);
  float* bv    = carve((size_t)Menc * Dm);
  float* bt0   = carve((size_t)Menc * Dm);
  float* bt1   = carve((size_t)Menc * Dm);
  float* bh    = carve((size_t)Menc * Dh);
  float* dec_x = carve((size_t)Mdec * Dm);

  auto gemm = [&](const float* A, const float* W, const float* bias, float* Cc,
                  int M, int N, int K, int relu) {
    dim3 g(N / TN, M / TM);
    k_gemm<<<g, dim3(256), 0, stream>>>(A, W, bias, Cc, M, N, K, relu);
  };

  // ================= encoder =================
  k_embed<<<Menc, 256, 0, stream>>>(src_seq, emb, nullptr, nullptr, enc_x, Ls);
  for (int i = 0; i < NL; i++) {
    const size_t DD = (size_t)Dm * Dm;
    gemm(enc_x, enc_qkv_w + (size_t)(i * 3 + 0) * DD, enc_qkv_b + (i * 3 + 0) * Dm, bq, Menc, Dm, Dm, 0);
    gemm(enc_x, enc_qkv_w + (size_t)(i * 3 + 1) * DD, enc_qkv_b + (i * 3 + 1) * Dm, bk, Menc, Dm, Dm, 0);
    gemm(enc_x, enc_qkv_w + (size_t)(i * 3 + 2) * DD, enc_qkv_b + (i * 3 + 2) * Dm, bv, Menc, Dm, Dm, 0);
    k_attn<<<Bb * Hh * Ls, 256, 0, stream>>>(bq, bk, bv, bt0, src_seq, Ls, Ls, 0);
    gemm(bt0, enc_fc_w + (size_t)i * DD, enc_fc_b + i * Dm, bt1, Menc, Dm, Dm, 0);
    k_res_ln<<<Menc, 256, 0, stream>>>(bt1, enc_x,
        enc_ln + (size_t)((i * 2 + 0) * 2 + 0) * Dm,
        enc_ln + (size_t)((i * 2 + 0) * 2 + 1) * Dm, src_seq, enc_x);
    gemm(enc_x, enc_ffn_w1 + (size_t)i * Dh * Dm, enc_ffn_b1 + i * Dh, bh, Menc, Dh, Dm, 1);
    gemm(bh, enc_ffn_w2 + (size_t)i * Dm * Dh, enc_ffn_b2 + i * Dm, bt1, Menc, Dm, Dh, 0);
    k_res_ln<<<Menc, 256, 0, stream>>>(bt1, enc_x,
        enc_ln + (size_t)((i * 2 + 1) * 2 + 0) * Dm,
        enc_ln + (size_t)((i * 2 + 1) * 2 + 1) * Dm, src_seq, enc_x);
  }

  // ================= decoder =================
  k_embed<<<Mdec, 256, 0, stream>>>(tgt_seq, emb, act_vecs, act_w, dec_x, Lt);
  for (int i = 0; i < NL; i++) {
    const size_t DD = (size_t)Dm * Dm;
    // ---- self-attention (causal) ----
    {
      const size_t wb = (size_t)((i * 2 + 0) * 3);
      gemm(dec_x, dec_qkv_w + (wb + 0) * DD, dec_qkv_b + (wb + 0) * Dm, bq, Mdec, Dm, Dm, 0);
      gemm(dec_x, dec_qkv_w + (wb + 1) * DD, dec_qkv_b + (wb + 1) * Dm, bk, Mdec, Dm, Dm, 0);
      gemm(dec_x, dec_qkv_w + (wb + 2) * DD, dec_qkv_b + (wb + 2) * Dm, bv, Mdec, Dm, Dm, 0);
      k_attn<<<Bb * Hh * Lt, 256, 0, stream>>>(bq, bk, bv, bt0, tgt_seq, Lt, Lt, 1);
      gemm(bt0, dec_fc_w + (size_t)(i * 2 + 0) * DD, dec_fc_b + (i * 2 + 0) * Dm, bt1, Mdec, Dm, Dm, 0);
      k_res_ln<<<Mdec, 256, 0, stream>>>(bt1, dec_x,
          dec_ln + (size_t)((i * 3 + 0) * 2 + 0) * Dm,
          dec_ln + (size_t)((i * 3 + 0) * 2 + 1) * Dm, tgt_seq, dec_x);
    }
    // ---- cross-attention (q from decoder, k/v from encoder output) ----
    {
      const size_t wb = (size_t)((i * 2 + 1) * 3);
      gemm(dec_x, dec_qkv_w + (wb + 0) * DD, dec_qkv_b + (wb + 0) * Dm, bq, Mdec, Dm, Dm, 0);
      gemm(enc_x, dec_qkv_w + (wb + 1) * DD, dec_qkv_b + (wb + 1) * Dm, bk, Menc, Dm, Dm, 0);
      gemm(enc_x, dec_qkv_w + (wb + 2) * DD, dec_qkv_b + (wb + 2) * Dm, bv, Menc, Dm, Dm, 0);
      k_attn<<<Bb * Hh * Lt, 256, 0, stream>>>(bq, bk, bv, bt0, src_seq, Lt, Ls, 0);
      gemm(bt0, dec_fc_w + (size_t)(i * 2 + 1) * DD, dec_fc_b + (i * 2 + 1) * Dm, bt1, Mdec, Dm, Dm, 0);
      k_res_ln<<<Mdec, 256, 0, stream>>>(bt1, dec_x,
          dec_ln + (size_t)((i * 3 + 1) * 2 + 0) * Dm,
          dec_ln + (size_t)((i * 3 + 1) * 2 + 1) * Dm, tgt_seq, dec_x);
    }
    // ---- FFN ----
    gemm(dec_x, dec_ffn_w1 + (size_t)i * Dh * Dm, dec_ffn_b1 + i * Dh, bh, Mdec, Dh, Dm, 1);
    gemm(bh, dec_ffn_w2 + (size_t)i * Dm * Dh, dec_ffn_b2 + i * Dm, bt1, Mdec, Dm, Dh, 0);
    k_res_ln<<<Mdec, 256, 0, stream>>>(bt1, dec_x,
        dec_ln + (size_t)((i * 3 + 2) * 2 + 0) * Dm,
        dec_ln + (size_t)((i * 3 + 2) * 2 + 1) * Dm, tgt_seq, dec_x);
  }

  // ================= logits =================
  gemm(dec_x, proj_w, nullptr, out, Mdec, Vv, Dm, 0);
}
